// GSN_45028437132005
// MI455X (gfx1250) — matrix operand order repaired and checked by reading, not compiled, not run on hardware
//
#include <hip/hip_runtime.h>
#include <hip/hip_bf16.h>
#include <stdint.h>

typedef __bf16 bf16;
typedef __attribute__((ext_vector_type(16))) __bf16 v16bf;
typedef __attribute__((ext_vector_type(8)))  float  v8f;

// ---------- helpers ----------

__device__ __forceinline__ bf16 f2bf(float x) {
  uint32_t u = __builtin_bit_cast(uint32_t, x);
  uint32_t r = (u + 0x7FFFu + ((u >> 16) & 1u)) >> 16;
  uint16_t h = (uint16_t)r;
  return __builtin_bit_cast(bf16, h);
}

__device__ __forceinline__ float sigf(float x) {
  return 1.0f / (1.0f + expf(-x));
}

union FragU { v16bf v; uint32_t u[8]; bf16 h[16]; };

// A fragment (16x32 bf16, row-major source, ldk elements per row)
// lane 0-15: M=lane,   K = k0+0..15
// lane16-31: M=lane-16,K = k0+16..31
__device__ __forceinline__ v16bf load_a_frag(const bf16* A, int ldk, int k0, int lane) {
  int r  = lane & 15;
  int ks = (lane >> 4) << 4;
  FragU f;
  const uint32_t* p0 = (const uint32_t*)(A + r * ldk + k0 + ks);
#pragma unroll
  for (int i = 0; i < 8; ++i) f.u[i] = p0[i];
  return f.v;
}

// B fragment (32x16 bf16) from weight matrix W[N][K] (row-major), computing x@W^T:
// B[k][n] = W[n][k].  lane 0-15: N=ncol+lane, K=k0+0..15 ; lane16-31: K=k0+16..31
__device__ __forceinline__ v16bf load_b_frag(const bf16* W, int ldk, int ncol, int k0, int lane) {
  int nn = ncol + (lane & 15);
  int kb = k0 + ((lane >> 4) << 4);
  FragU f;
  const uint32_t* p = (const uint32_t*)(W + (size_t)nn * ldk + kb);
#pragma unroll
  for (int i = 0; i < 8; ++i) f.u[i] = p[i];
  return f.v;
}

__device__ __forceinline__ v8f wmma_bf16(v16bf a, v16bf b, v8f c) {
  return __builtin_amdgcn_wmma_f32_16x16x32_bf16(false, a, false, b, (short)0, c, false, false);
}

// monotone float <-> uint mapping for atomic max over signed floats
__device__ __forceinline__ unsigned int fmap(float f) {
  uint32_t b = __builtin_bit_cast(uint32_t, f);
  return b ^ ((uint32_t)(((int32_t)b) >> 31) | 0x80000000u);
}
__device__ __forceinline__ float funmap(unsigned int u) {
  uint32_t b = (u & 0x80000000u) ? (u ^ 0x80000000u) : ~u;
  return __builtin_bit_cast(float, b);
}

// ---------- conversion: f32 -> bf16 with optional K padding ----------
__global__ void k_conv(const float* __restrict__ src, bf16* __restrict__ dst,
                       int rows, int Ks, int Kd) {
  int idx = blockIdx.x * blockDim.x + threadIdx.x;
  if (idx >= rows * Kd) return;
  int r = idx / Kd, c = idx - r * Kd;
  float v = (c < Ks) ? src[(size_t)r * Ks + c] : 0.0f;
  dst[idx] = f2bf(v);
}

// ---------- h = x @ lin_W^T + lin_b ----------
__global__ void k_lin(const bf16* __restrict__ xb, const bf16* __restrict__ wlin,
                      const float* __restrict__ linb,
                      float* __restrict__ h, bf16* __restrict__ hb, int nN) {
  int node0 = blockIdx.x * 16;
  int w = threadIdx.x >> 5, lane = threadIdx.x & 31;
  int nc = w * 16;
  const bf16* A = xb + (size_t)node0 * 64;
  v8f acc = {};
#pragma unroll
  for (int k0 = 0; k0 < 64; k0 += 32) {
    v16bf a = load_a_frag(A, 64, k0, lane);
    v16bf b = load_b_frag(wlin, 64, nc, k0, lane);
    acc = wmma_bf16(a, b, acc);
  }
  int nl = nc + (lane & 15);
  float bias = linb[nl];
  int mbase = (lane >> 4) << 3;
  if (node0 + 16 <= nN) {   // full block: uniform, unguarded epilogue
#pragma unroll
    for (int e = 0; e < 8; ++e) {
      size_t off = (size_t)(node0 + mbase + e) * 128 + nl;
      float val = acc[e] + bias;
      h[off] = val;
      hb[off] = f2bf(val);
    }
  } else {
#pragma unroll
    for (int e = 0; e < 8; ++e) {
      int node = node0 + mbase + e;
      if (node < nN) {
        float val = acc[e] + bias;
        h [(size_t)node * 128 + nl] = val;
        hb[(size_t)node * 128 + nl] = f2bf(val);
      }
    }
  }
}

// ---------- fused edge message MLP + weighted scatter-add ----------
// 32 edges per block; A = [h[ni] | h[no] | struct[ni] | struct[no] | ef | pad] (K=288)
__global__ void k_msg(const int* __restrict__ el, const float* __restrict__ ef,
                      const int* __restrict__ ns, const float* __restrict__ ew,
                      const bf16* __restrict__ hb,
                      const bf16* __restrict__ w1, const float* __restrict__ b1,
                      const bf16* __restrict__ w2, const float* __restrict__ b2,
                      float* __restrict__ upd, int nE) {
  __shared__ bf16 As[32 * 288];
  __shared__ bf16 M1[32 * 128];
  __shared__ int   NIS[32];
  __shared__ int   NOS[32];
  __shared__ float EWS[32];
  int edge0 = blockIdx.x * 32;
  int tid = threadIdx.x;

  // stage per-edge metadata once
  if (tid < 32) {
    int e = edge0 + tid; if (e >= nE) e = nE - 1;
    NIS[tid] = el[2 * e];
    NOS[tid] = el[2 * e + 1];
    EWS[tid] = ew[e];
    if (edge0 + 32 + tid < nE)
      __builtin_prefetch(el + 2 * (edge0 + 32 + tid), 0, 0);
  }
  __syncthreads();

  for (int idx = tid; idx < 32 * 288; idx += 256) {
    int r = idx / 288;
    int c = idx - r * 288;
    int e = edge0 + r; if (e >= nE) e = nE - 1;
    int ni = NIS[r], no = NOS[r];
    bf16 val;
    if      (c < 128) val = hb[(size_t)ni * 128 + c];
    else if (c < 256) val = hb[(size_t)no * 128 + (c - 128)];
    else if (c < 262) val = f2bf((float)ns[ni * 6 + (c - 256)]);
    else if (c < 268) val = f2bf((float)ns[no * 6 + (c - 262)]);
    else if (c < 284) val = f2bf(ef[(size_t)e * 16 + (c - 268)]);
    else              val = f2bf(0.0f);
    As[idx] = val;
  }
  __syncthreads();

  int w = tid >> 5, lane = tid & 31;
  int nc = w * 16;
  int nl = nc + (lane & 15);
  int mbase = (lane >> 4) << 3;

  // m1 = relu(A @ W1^T + b1)   (K=288)
#pragma unroll
  for (int mt = 0; mt < 2; ++mt) {
    v8f acc = {};
    for (int k0 = 0; k0 < 288; k0 += 32) {
      v16bf a = load_a_frag(As + mt * 16 * 288, 288, k0, lane);
      v16bf b = load_b_frag(w1, 288, nc, k0, lane);
      acc = wmma_bf16(a, b, acc);
    }
    float bias = b1[nl];
#pragma unroll
    for (int e = 0; e < 8; ++e) {
      float val = acc[e] + bias;
      if (val < 0.f) val = 0.f;
      M1[(mt * 16 + mbase + e) * 128 + nl] = f2bf(val);
    }
  }
  __syncthreads();

  // m2 = m1 @ W2^T + b2 ; upd[no] += ew * m2   (K=128)
  bool full = (edge0 + 32 <= nE);
#pragma unroll
  for (int mt = 0; mt < 2; ++mt) {
    v8f acc = {};
#pragma unroll
    for (int k0 = 0; k0 < 128; k0 += 32) {
      v16bf a = load_a_frag(M1 + mt * 16 * 128, 128, k0, lane);
      v16bf b = load_b_frag(w2, 128, nc, k0, lane);
      acc = wmma_bf16(a, b, acc);
    }
    float bias = b2[nl];
    if (full) {   // uniform fast path
#pragma unroll
      for (int e8 = 0; e8 < 8; ++e8) {
        int r = mt * 16 + mbase + e8;
        float val = (acc[e8] + bias) * EWS[r];
        atomicAdd(&upd[(size_t)NOS[r] * 128 + nl], val);
      }
    } else {
#pragma unroll
      for (int e8 = 0; e8 < 8; ++e8) {
        int r = mt * 16 + mbase + e8;
        if (edge0 + r < nE) {
          float val = (acc[e8] + bias) * EWS[r];
          atomicAdd(&upd[(size_t)NOS[r] * 128 + nl], val);
        }
      }
    }
  }
}

// ---------- fused node update MLP + GRU ----------
__global__ void k_node(const float* __restrict__ updv,
                       float* __restrict__ h, bf16* __restrict__ hb,
                       const bf16* __restrict__ w1, const float* __restrict__ b1,
                       const bf16* __restrict__ w2, const float* __restrict__ b2,
                       const bf16* __restrict__ wgi, const bf16* __restrict__ wgh,
                       const float* __restrict__ gbi, const float* __restrict__ gbh,
                       int nN) {
  __shared__ bf16 CIN[16 * 256];
  __shared__ bf16 MID[16 * 128];
  __shared__ bf16 OUT[16 * 128];
  __shared__ bf16 HBS[16 * 128];
  int node0 = blockIdx.x * 16;
  int tid = threadIdx.x;

  for (int idx = tid; idx < 16 * 256; idx += 256) {
    int r = idx >> 8, c = idx & 255;
    int node = node0 + r; if (node >= nN) node = nN - 1;
    if (c < 128) {
      bf16 v = hb[(size_t)node * 128 + c];
      CIN[idx] = v;
      HBS[r * 128 + c] = v;
    } else {
      CIN[idx] = f2bf(updv[(size_t)node * 128 + (c - 128)]);
    }
  }
  __syncthreads();

  int w = tid >> 5, lane = tid & 31;
  int nc = w * 16;
  int nl = nc + (lane & 15);
  int mbase = (lane >> 4) << 3;

  { // mid = relu([h|upd] @ upd_W1^T + b1)   (K=256)
    v8f acc = {};
#pragma unroll
    for (int k0 = 0; k0 < 256; k0 += 32) {
      v16bf a = load_a_frag(CIN, 256, k0, lane);
      v16bf b = load_b_frag(w1, 256, nc, k0, lane);
      acc = wmma_bf16(a, b, acc);
    }
    float bias = b1[nl];
#pragma unroll
    for (int e = 0; e < 8; ++e) {
      float val = acc[e] + bias; if (val < 0.f) val = 0.f;
      MID[(mbase + e) * 128 + nl] = f2bf(val);
    }
  }
  __syncthreads();
  { // out = relu(mid @ upd_W2^T + b2)  (outer relu folded, K=128)
    v8f acc = {};
#pragma unroll
    for (int k0 = 0; k0 < 128; k0 += 32) {
      v16bf a = load_a_frag(MID, 128, k0, lane);
      v16bf b = load_b_frag(w2, 128, nc, k0, lane);
      acc = wmma_bf16(a, b, acc);
    }
    float bias = b2[nl];
#pragma unroll
    for (int e = 0; e < 8; ++e) {
      float val = acc[e] + bias; if (val < 0.f) val = 0.f;
      OUT[(mbase + e) * 128 + nl] = f2bf(val);
    }
  }
  __syncthreads();

  // GRU gates: wave owns column nl across all 3 gates of gi and gh
  v8f gi[3], gh[3];
#pragma unroll
  for (int gidx = 0; gidx < 3; ++gidx) {
    v8f ai = {}, ah = {};
#pragma unroll
    for (int k0 = 0; k0 < 128; k0 += 32) {
      v16bf ao = load_a_frag(OUT, 128, k0, lane);
      v16bf bI = load_b_frag(wgi, 128, gidx * 128 + nc, k0, lane);
      ai = wmma_bf16(ao, bI, ai);
      v16bf ax = load_a_frag(HBS, 128, k0, lane);
      v16bf bH = load_b_frag(wgh, 128, gidx * 128 + nc, k0, lane);
      ah = wmma_bf16(ax, bH, ah);
    }
    gi[gidx] = ai; gh[gidx] = ah;
  }
  float bir = gbi[nl], biz = gbi[128 + nl], bin = gbi[256 + nl];
  float bhr = gbh[nl], bhz = gbh[128 + nl], bhn = gbh[256 + nl];
  bool fullb = (node0 + 16 <= nN);
#pragma unroll
  for (int e = 0; e < 8; ++e) {
    int node = node0 + mbase + e;
    if (fullb || node < nN) {
      float ir = gi[0][e] + bir, iz = gi[1][e] + biz, inn = gi[2][e] + bin;
      float hr = gh[0][e] + bhr, hz = gh[1][e] + bhz, hn  = gh[2][e] + bhn;
      float r = sigf(ir + hr);
      float z = sigf(iz + hz);
      float ng = tanhf(inn + r * hn);
      size_t off = (size_t)node * 128 + nl;
      float hold = h[off];
      float hnew = (1.f - z) * ng + z * hold;
      h[off]  = hnew;
      hb[off] = f2bf(hnew);
    }
  }
}

// ---------- Set2Set LSTM step (8 graphs, one block) ----------
__global__ void k_lstm(const float* __restrict__ Wi, const float* __restrict__ Wh,
                       const float* __restrict__ bi, const float* __restrict__ bh,
                       float* __restrict__ hg, float* __restrict__ cg,
                       float* __restrict__ rf, float* __restrict__ den,
                       unsigned int* __restrict__ gmax) {
  __shared__ float g4[8 * 512];
  int tid = threadIdx.x;
  for (int idx = tid; idx < 8 * 512; idx += 256) {
    int g = idx >> 9, j = idx & 511;
    float acc = bi[j] + bh[j];
    const float* wi  = Wi + (size_t)j * 256;
    const float* wh  = Wh + (size_t)j * 128;
    const float* hgg = hg + g * 128;   // q_star[:, :128] == previous hg
    const float* rfg = rf + g * 128;   // q_star[:, 128:]
    for (int k = 0; k < 128; ++k)
      acc += hgg[k] * (wi[k] + wh[k]) + rfg[k] * wi[128 + k];
    g4[idx] = acc;
  }
  __syncthreads();
  for (int idx = tid; idx < 8 * 128; idx += 256) {
    int g = idx >> 7, k = idx & 127;
    const float* gg = g4 + g * 512;
    float c = sigf(gg[128 + k]) * cg[idx] + sigf(gg[k]) * tanhf(gg[256 + k]);
    cg[idx] = c;
    hg[idx] = sigf(gg[384 + k]) * tanhf(c);
  }
  __syncthreads();
  for (int idx = tid; idx < 8 * 128; idx += 256) rf[idx] = 0.f; // reset for attention
  if (tid < 8) { den[tid] = 0.f; gmax[tid] = 0u; }
}

// ---------- attention: e = <nf_i, q_g>, segment max ----------
__global__ void k_att1(const float* __restrict__ nf, const float* __restrict__ hg,
                       const int* __restrict__ n2g, float* __restrict__ ebuf,
                       unsigned int* __restrict__ gmax, int nN) {
  int i = blockIdx.x * blockDim.x + threadIdx.x;
  if (i >= nN) return;
  int g = n2g[i];
  const float* row = nf + (size_t)i * 128;
  const float* q = hg + g * 128;
  float acc = 0.f;
  for (int k = 0; k < 128; ++k) acc += row[k] * q[k];
  ebuf[i] = acc;
  atomicMax(&gmax[g], fmap(acc));
}

// ---------- a = exp(e - max), segment sum ----------
__global__ void k_att2(float* __restrict__ ebuf, const unsigned int* __restrict__ gmax,
                       const int* __restrict__ n2g, float* __restrict__ den, int nN) {
  int i = blockIdx.x * blockDim.x + threadIdx.x;
  if (i >= nN) return;
  int g = n2g[i];
  float a = expf(ebuf[i] - funmap(gmax[g]));
  ebuf[i] = a;
  atomicAdd(&den[g], a);
}

// ---------- rf[g] += sum_i att_i * nf_i  (LDS-reduced per block) ----------
__global__ void k_att3(const float* __restrict__ nf, const float* __restrict__ abuf,
                       const float* __restrict__ den, const int* __restrict__ n2g,
                       float* __restrict__ rf, int nN) {
  __shared__ float accs[8 * 128];
  int tid = threadIdx.x;
  int node0 = blockIdx.x * 256;
  for (int idx = tid; idx < 1024; idx += 256) accs[idx] = 0.f;
  __syncthreads();
  for (int t = tid; t < 256 * 128; t += 256) {
    int r = t >> 7, c = t & 127;
    int i = node0 + r;
    if (i < nN) {
      int g = n2g[i];
      float att = abuf[i] / den[g];
      atomicAdd(&accs[g * 128 + c], att * nf[(size_t)i * 128 + c]);
    }
  }
  __syncthreads();
  for (int idx = tid; idx < 1024; idx += 256)
    atomicAdd(&rf[idx], accs[idx]);
}

// ---------- graph_feature = concat(hg, rf) ----------
__global__ void k_final(const float* __restrict__ hg, const float* __restrict__ rf,
                        float* __restrict__ out) {
  int idx = blockIdx.x * blockDim.x + threadIdx.x;
  if (idx >= 2048) return;
  int g = idx >> 8, c = idx & 255;
  out[idx] = (c < 128) ? hg[g * 128 + c] : rf[g * 128 + (c - 128)];
}

// ---------- host ----------
extern "C" void kernel_launch(void* const* d_in, const int* in_sizes, int n_in,
                              void* d_out, int out_size, void* d_ws, size_t ws_size,
                              hipStream_t stream) {
  (void)n_in; (void)out_size; (void)ws_size;
  const float* x    = (const float*)d_in[0];
  const int*   el   = (const int*)  d_in[1];
  const float* ef   = (const float*)d_in[2];
  const int*   ns   = (const int*)  d_in[3];
  const float* ew   = (const float*)d_in[4];
  const int*   n2g  = (const int*)  d_in[5];
  const float* lin_W = (const float*)d_in[6];
  const float* lin_b = (const float*)d_in[7];
  const float* msg_W1= (const float*)d_in[8];
  const float* msg_b1= (const float*)d_in[9];
  const float* msg_W2= (const float*)d_in[10];
  const float* msg_b2= (const float*)d_in[11];
  const float* upd_W1= (const float*)d_in[12];
  const float* upd_b1= (const float*)d_in[13];
  const float* upd_W2= (const float*)d_in[14];
  const float* upd_b2= (const float*)d_in[15];
  const float* gru_Wi= (const float*)d_in[16];
  const float* gru_Wh= (const float*)d_in[17];
  const float* gru_bi= (const float*)d_in[18];
  const float* gru_bh= (const float*)d_in[19];
  const float* lstm_Wi=(const float*)d_in[20];
  const float* lstm_Wh=(const float*)d_in[21];
  const float* lstm_bi=(const float*)d_in[22];
  const float* lstm_bh=(const float*)d_in[23];

  int nN = in_sizes[0] / 64;
  int nE = in_sizes[1] / 2;

  char* p = (char*)d_ws;
  auto take = [&](size_t bytes) {
    char* r = p; p += (bytes + 255) & ~(size_t)255; return r;
  };
  bf16* wlin  = (bf16*)take(128u * 64 * 2);
  bf16* wmsg1 = (bf16*)take(128u * 288 * 2);
  bf16* wmsg2 = (bf16*)take(128u * 128 * 2);
  bf16* wupd1 = (bf16*)take(128u * 256 * 2);
  bf16* wupd2 = (bf16*)take(128u * 128 * 2);
  bf16* wgi   = (bf16*)take(384u * 128 * 2);
  bf16* wgh   = (bf16*)take(384u * 128 * 2);
  bf16* xb    = (bf16*)take((size_t)nN * 64 * 2);
  float* h    = (float*)take((size_t)nN * 128 * 4);
  bf16* hb    = (bf16*)take((size_t)nN * 128 * 2);
  float* upd  = (float*)take((size_t)nN * 128 * 4);
  float* hg   = (float*)take(8u * 128 * 4);   // hg, cg, rf contiguous (4096 each)
  float* cg   = (float*)take(8u * 128 * 4);
  float* rf   = (float*)take(8u * 128 * 4);
  float* den  = (float*)take(8u * 4);
  unsigned int* gmax = (unsigned int*)take(8u * 4);
  float* ebuf = (float*)take((size_t)nN * 4);
  (void)cg;

  auto conv = [&](const float* s, bf16* d, int rows, int Ks, int Kd) {
    int total = rows * Kd;
    k_conv<<<(total + 255) / 256, 256, 0, stream>>>(s, d, rows, Ks, Kd);
  };
  conv(lin_W,  wlin,  128, 64, 64);
  conv(msg_W1, wmsg1, 128, 284, 288);
  conv(msg_W2, wmsg2, 128, 128, 128);
  conv(upd_W1, wupd1, 128, 256, 256);
  conv(upd_W2, wupd2, 128, 128, 128);
  conv(gru_Wi, wgi,   384, 128, 128);
  conv(gru_Wh, wgh,   384, 128, 128);
  conv(x,      xb,    nN,  64, 64);

  int nb16 = (nN + 15) / 16;
  k_lin<<<nb16, 256, 0, stream>>>(xb, wlin, lin_b, h, hb, nN);

  for (int layer = 0; layer < 2; ++layer) {
    hipMemsetAsync(upd, 0, (size_t)nN * 128 * 4, stream);
    k_msg<<<(nE + 31) / 32, 256, 0, stream>>>(el, ef, ns, ew, hb,
                                              wmsg1, msg_b1, wmsg2, msg_b2,
                                              upd, nE);
    k_node<<<nb16, 256, 0, stream>>>(upd, h, hb,
                                     wupd1, upd_b1, wupd2, upd_b2,
                                     wgi, wgh, gru_bi, gru_bh, nN);
  }

  // Set2Set readout
  hipMemsetAsync(hg, 0, 3u * 4096, stream);  // hg, cg, rf
  int nb256 = (nN + 255) / 256;
  for (int it = 0; it < 3; ++it) {
    k_lstm<<<1, 256, 0, stream>>>(lstm_Wi, lstm_Wh, lstm_bi, lstm_bh,
                                  hg, cg, rf, den, gmax);
    k_att1<<<nb256, 256, 0, stream>>>(h, hg, n2g, ebuf, gmax, nN);
    k_att2<<<nb256, 256, 0, stream>>>(ebuf, gmax, n2g, den, nN);
    k_att3<<<nb256, 256, 0, stream>>>(h, ebuf, den, n2g, rf, nN);
  }

  k_final<<<8, 256, 0, stream>>>(hg, rf, (float*)d_out);
  hipMemcpyAsync((float*)d_out + 2048, h, (size_t)nN * 128 * 4,
                 hipMemcpyDeviceToDevice, stream);
}
